// EQL_Network_31877247270990
// MI455X (gfx1250) — matrix operand order repaired and checked by reading, not compiled, not run on hardware
//
#include <hip/hip_runtime.h>
#include <hip/hip_bf16.h>
#include <math.h>
#include <stdint.h>

// ---------------------------------------------------------------------------
// EQL network forward on MI455X (gfx1250), wave32.
//   - v_wmma_f32_16x16x32_f16 matrix core (f16 data, f32 accum)
//   - global_load_async_to_lds_b128 (ASYNCcnt-tracked DMA) for tile staging
// GEMM chain: 8192 x [160 -> 2112 -> 4224 -> 8448 -> 4224 -> 64]
// Weights are converted fp32->f16 once per launch (each layer's weights fit
// in the 192MB L2, re-reads across the 64 M-tiles stay on-chip).
// ---------------------------------------------------------------------------

typedef __attribute__((ext_vector_type(16))) _Float16 v16h;
typedef __attribute__((ext_vector_type(8)))  float    v8f;

#define TM 128
#define TN 128
#define TK 32
#define LDSS 40   // LDS row stride in halves (80B = 5*16B): aligned + conflict-free

union V8FU { v8f v; float f[8]; };

// A fragment (16x32, MxK): lane L holds row M=L&15.
// lanes 0-15: K {0..7, 16..23}; lanes 16-31: K {8..15, 24..31}.
__device__ __forceinline__ v16h fragA(const _Float16* row, int lane) {
    const int kb = (lane < 16) ? 0 : 8;
    v16h f;
    ((uint4*)&f)[0] = *(const uint4*)(row + kb);
    ((uint4*)&f)[1] = *(const uint4*)(row + kb + 16);
    return f;
}

// B fragment (32x16, KxN): lane L holds column N=L&15 (== a row of W).
// lanes 0-15: K {0..7, 16..23}; lanes 16-31: K {8..15, 24..31}.
__device__ __forceinline__ v16h fragB(const _Float16* row, int lane) {
    const int kb = (lane < 16) ? 0 : 8;
    v16h f;
    ((uint4*)&f)[0] = *(const uint4*)(row + kb);
    ((uint4*)&f)[1] = *(const uint4*)(row + kb + 16);
    return f;
}

// Async DMA: 16 bytes / lane, global -> LDS, tracked by ASYNCcnt.
__device__ __forceinline__ void async_b128(const void* g, void* l) {
    asm volatile("global_load_async_to_lds_b128 %0, %1, off"
                 :: "v"((uint32_t)(uintptr_t)l), "v"((uint64_t)(uintptr_t)g)
                 : "memory");
}

__global__ __launch_bounds__(256)
void gemm_wmma_f16(const _Float16* __restrict__ A,   // [M,K] f16 (K mult of 32)
                   const _Float16* __restrict__ Wh,  // [N,K] f16 row-major
                   const float*    __restrict__ bias,// [N]
                   _Float16*       __restrict__ outH,// [M,N] f16 or null
                   float*          __restrict__ outF,// [M,N] f32 or null
                   int M, int N, int K, int do_relu)
{
    __shared__ _Float16 sA[2][TM][LDSS];
    __shared__ _Float16 sB[2][TN][LDSS];

    const int tid   = threadIdx.x;
    const int lane  = tid & 31;
    const int wave  = tid >> 5;
    const int waveM = wave >> 1;   // 0..3 -> 32-row slab
    const int waveN = wave & 1;    // 0..1 -> 64-col slab
    const int blockM = blockIdx.y * TM;
    const int blockN = blockIdx.x * TN;

    // Issue one 128x32-half tile pair via async DMA (4 async ops per wave,
    // uniform across all waves -> ASYNCcnt bookkeeping is exact).
    auto issueTile = [&](int s) {
        const int buf = s & 1;
        const int k0  = s * TK;
        #pragma unroll
        for (int i = 0; i < 2; ++i) {          // A tile: 512 chunks of 8 halves
            int c   = tid + i * 256;
            int row = c >> 2;
            int col = (c & 3) * 8;
            async_b128(A + (size_t)(blockM + row) * K + (k0 + col),
                       &sA[buf][row][col]);
        }
        #pragma unroll
        for (int i = 0; i < 2; ++i) {          // B tile (rows of W)
            int c   = tid + i * 256;
            int row = c >> 2;
            int col = (c & 3) * 8;
            int n   = blockN + row;
            if (n >= N) n = N - 1;             // clamp: cols >= N never stored
            async_b128(Wh + (size_t)n * K + (k0 + col),
                       &sB[buf][row][col]);
        }
    };

    v8f acc[2][4];
    const v8f vzero = {0.f,0.f,0.f,0.f,0.f,0.f,0.f,0.f};
    #pragma unroll
    for (int mi = 0; mi < 2; ++mi)
        #pragma unroll
        for (int ni = 0; ni < 4; ++ni) acc[mi][ni] = vzero;

    const int nSteps = K / TK;
    issueTile(0);
    if (nSteps > 1) issueTile(1);

    for (int s = 0; s < nSteps; ++s) {
        const int buf = s & 1;
        // own group s done; group s+1 (4 ops) may remain in flight
        if (s + 1 < nSteps) asm volatile("s_wait_asynccnt 4" ::: "memory");
        else                asm volatile("s_wait_asynccnt 0" ::: "memory");
        __syncthreads();                       // everyone's tile s is in LDS

        v16h af[2], bf[4];
        #pragma unroll
        for (int mi = 0; mi < 2; ++mi)
            af[mi] = fragA(&sA[buf][waveM * 32 + mi * 16 + (lane & 15)][0], lane);
        #pragma unroll
        for (int ni = 0; ni < 4; ++ni)
            bf[ni] = fragB(&sB[buf][waveN * 64 + ni * 16 + (lane & 15)][0], lane);

        #pragma unroll
        for (int mi = 0; mi < 2; ++mi)
            #pragma unroll
            for (int ni = 0; ni < 4; ++ni)
                acc[mi][ni] = __builtin_amdgcn_wmma_f32_16x16x32_f16(
                    false, af[mi], false, bf[ni],
                    (short)0, acc[mi][ni], false, false);

        __syncthreads();                       // all reads of buf done
        if (s + 2 < nSteps) issueTile(s + 2);  // safe to overwrite buf now
    }

    // ---- epilogue: bias + relu, write f16 activation and/or f32 out ----
    #pragma unroll
    for (int ni = 0; ni < 4; ++ni) {
        const int n = blockN + waveN * 64 + ni * 16 + (lane & 15);
        const float bn = (n < N) ? bias[n] : 0.f;
        #pragma unroll
        for (int mi = 0; mi < 2; ++mi) {
            const int mBase = blockM + waveM * 32 + mi * 16 + ((lane >> 4) << 3);
            V8FU u; u.v = acc[mi][ni];
            #pragma unroll
            for (int r = 0; r < 8; ++r) {
                float v = u.f[r] + bn;
                if (do_relu) v = fmaxf(v, 0.f);
                if (n < N) {
                    const size_t off = (size_t)(mBase + r) * N + n;
                    if (outH) outH[off] = (_Float16)v;
                    if (outF) outF[off] = v;
                }
            }
        }
    }
}

// fp32 [N,Kw] -> f16 [N,Kp], zero-padded in K (one-shot per launch).
__global__ void cvt_weight(const float* __restrict__ W, _Float16* __restrict__ Wh,
                           int N, int Kw, int Kp)
{
    int idx = blockIdx.x * blockDim.x + threadIdx.x;
    if (idx >= N * Kp) return;
    int n = idx / Kp, k = idx - n * Kp;
    Wh[idx] = (k < Kw) ? (_Float16)W[(size_t)n * Kw + k] : (_Float16)0.f;
}

// Build x = concat(state, preference) as f16, K padded 132 -> 160 with zeros.
__global__ void prep_x(const float* __restrict__ state,
                       const float* __restrict__ pref,
                       _Float16* __restrict__ x, int rows)
{
    int idx = blockIdx.x * blockDim.x + threadIdx.x;
    if (idx >= rows * 160) return;
    int r = idx / 160, c = idx - r * 160;
    float v = 0.f;
    if (c < 128)       v = state[r * 128 + c];
    else if (c < 132)  v = pref[r * 4 + (c - 128)];
    x[idx] = (_Float16)v;
}

// H(): per row, a* = argmax_a <q[row,a,:], pref[row,:]> (first-tie), hq = q[row,a*,:]
__global__ void hq_kernel(const float* __restrict__ q,    // [rows,16,4]
                          const float* __restrict__ pref, // [rows,4]
                          float* __restrict__ hq, int rows)
{
    int r = blockIdx.x * blockDim.x + threadIdx.x;
    if (r >= rows) return;
    const float* qr = q + (size_t)r * 64;
    const float p0 = pref[r*4+0], p1 = pref[r*4+1], p2 = pref[r*4+2], p3 = pref[r*4+3];
    float best = -INFINITY; int bi = 0;
    #pragma unroll
    for (int a = 0; a < 16; ++a) {
        float ip = qr[a*4+0]*p0 + qr[a*4+1]*p1 + qr[a*4+2]*p2 + qr[a*4+3]*p3;
        if (ip > best) { best = ip; bi = a; }
    }
    hq[r*4+0] = qr[bi*4+0];
    hq[r*4+1] = qr[bi*4+1];
    hq[r*4+2] = qr[bi*4+2];
    hq[r*4+3] = qr[bi*4+3];
}

extern "C" void kernel_launch(void* const* d_in, const int* in_sizes, int n_in,
                              void* d_out, int out_size, void* d_ws, size_t ws_size,
                              hipStream_t stream)
{
    const int B_ = 8192;
    const float* state = (const float*)d_in[0];
    const float* pref  = (const float*)d_in[1];
    const float* w1 = (const float*)d_in[2];  const float* b1 = (const float*)d_in[3];
    const float* w2 = (const float*)d_in[4];  const float* b2 = (const float*)d_in[5];
    const float* w3 = (const float*)d_in[6];  const float* b3 = (const float*)d_in[7];
    const float* w4 = (const float*)d_in[8];  const float* b4 = (const float*)d_in[9];
    const float* wq = (const float*)d_in[10]; const float* bq = (const float*)d_in[11];

    float* hq = (float*)d_out;              // [8192,4]
    float* q  = (float*)d_out + B_ * 4;     // [8192,64]

    // Workspace layout (~372 MB total):
    //   x16  : 8192*160 f16            =   2.6 MB
    //   bufA : 8192*8448 f16 (h1/h3)   = 138.4 MB
    //   bufB : 8192*4224 f16 (h2/h4)   =  69.2 MB
    //   w1h  : 2112*160  f16 (padded)  =   0.7 MB
    //   w2h  : 4224*2112 f16           =  17.8 MB
    //   w3h  : 8448*4224 f16           =  71.4 MB
    //   w4h  : 4224*8448 f16           =  71.4 MB
    //   wqh  : 64*4224   f16           =   0.5 MB
    char* ws = (char*)d_ws;
    size_t off = 0;
    auto take = [&](size_t halves) {
        _Float16* p = (_Float16*)(ws + off);
        off += halves * sizeof(_Float16);
        off = (off + 255) & ~(size_t)255;
        return p;
    };
    _Float16* x16  = take((size_t)B_ * 160);
    _Float16* bufA = take((size_t)B_ * 8448);
    _Float16* bufB = take((size_t)B_ * 4224);
    _Float16* w1h  = take((size_t)2112 * 160);
    _Float16* w2h  = take((size_t)4224 * 2112);
    _Float16* w3h  = take((size_t)8448 * 4224);
    _Float16* w4h  = take((size_t)4224 * 8448);
    _Float16* wqh  = take((size_t)64 * 4224);

    auto cvt = [&](const float* W, _Float16* Wh, int N, int Kw, int Kp) {
        int total = N * Kp;
        cvt_weight<<<(total + 255) / 256, 256, 0, stream>>>(W, Wh, N, Kw, Kp);
    };
    prep_x<<<(B_ * 160 + 255) / 256, 256, 0, stream>>>(state, pref, x16, B_);
    cvt(w1, w1h, 2112, 132,  160);
    cvt(w2, w2h, 4224, 2112, 2112);
    cvt(w3, w3h, 8448, 4224, 4224);
    cvt(w4, w4h, 4224, 8448, 8448);
    cvt(wq, wqh, 64,   4224, 4224);

    auto launch_gemm = [&](const _Float16* A, const _Float16* Wh, const float* bias,
                           _Float16* oH, float* oF, int N, int K, int relu) {
        dim3 grid((N + TN - 1) / TN, B_ / TM);
        gemm_wmma_f16<<<grid, 256, 0, stream>>>(A, Wh, bias, oH, oF, B_, N, K, relu);
    };

    launch_gemm(x16,  w1h, b1, bufA, nullptr, 2112, 160,  1); // h1
    launch_gemm(bufA, w2h, b2, bufB, nullptr, 4224, 2112, 1); // h2
    launch_gemm(bufB, w3h, b3, bufA, nullptr, 8448, 4224, 1); // h3
    launch_gemm(bufA, w4h, b4, bufB, nullptr, 4224, 8448, 1); // h4
    launch_gemm(bufB, wqh, bq, nullptr, q,      64, 4224, 0); // q (fp32 out)

    hq_kernel<<<(B_ + 255) / 256, 256, 0, stream>>>(q, pref, hq, B_);
}
